// SimpleRNN_64158221468095
// MI455X (gfx1250) — matrix operand order repaired and checked by reading, not compiled, not run on hardware
//
#include <hip/hip_runtime.h>
#include <hip/hip_bf16.h>

// Problem constants (match reference)
#define BB 64
#define TT 512
#define DIN 512
#define HH 1024
#define DOUT 512
#define THREADS 256

typedef __attribute__((ext_vector_type(8)))  __bf16 v8bf;
typedef __attribute__((ext_vector_type(16))) __bf16 v16bf;
typedef __attribute__((ext_vector_type(8)))  float  v8f;

__device__ __forceinline__ unsigned short f32_to_bf16_bits(float f) {
  unsigned int u = __float_as_uint(f);
  u += 0x7FFFu + ((u >> 16) & 1u);     // round-to-nearest-even
  return (unsigned short)(u >> 16);
}

// Build a v16bf operand from two 16-byte chunks.
__device__ __forceinline__ v16bf load_pair(const unsigned short* p0, const unsigned short* p1) {
  v8bf lo = *(const v8bf*)p0;
  v8bf hi = *(const v8bf*)p1;
  v16bf r;
#pragma unroll
  for (int i = 0; i < 8; ++i) { r[i] = lo[i]; r[i + 8] = hi[i]; }
  return r;
}

// ---- CDNA5 async global -> LDS copy (ASYNCcnt-tracked, no VGPR data path).
// lds_off: byte address in LDS (low 32 bits of a generic shared pointer);
// gaddr:   64-bit global address.
__device__ __forceinline__ void async_b128_to_lds(unsigned int lds_off, const void* gaddr) {
  asm volatile("global_load_async_to_lds_b128 %0, %1, off"
               :: "v"(lds_off), "v"(gaddr) : "memory");
}
__device__ __forceinline__ void wait_async0() {
  asm volatile("s_wait_asynccnt 0" ::: "memory");
}

// ---------------- weight packing: f32 (N x K row-major) -> bf16 B-operand tiles
// Tile (ntile, ktile) of 32x16 (KxN): lane l holds N = ntile*16 + (l&15),
// K = ktile*32 + (l<16 ? 0..15 : 16..31); 16 contiguous bf16 per lane.
__global__ void pack_weight_bf16(const float* __restrict__ W, unsigned short* __restrict__ out,
                                 int N, int K) {
  int wave = (blockIdx.x * blockDim.x + threadIdx.x) >> 5;
  int lane = threadIdx.x & 31;
  int KT = K >> 5, NT = N >> 4;
  if (wave >= NT * KT) return;
  int ntile = wave / KT;
  int ktile = wave - ntile * KT;
  int n = ntile * 16 + (lane & 15);
  int kbase = ktile * 32 + ((lane & 16) ? 16 : 0);
  unsigned short* dst = out + (((size_t)wave * 32) + lane) * 16;
  const float* src = W + (size_t)n * K + kbase;
#pragma unroll
  for (int j = 0; j < 16; ++j) dst[j] = f32_to_bf16_bits(src[j]);
}

// ---------------- elementwise f32 -> bf16
__global__ void cvt_f32_bf16(const float* __restrict__ in, unsigned short* __restrict__ out, int n) {
  int i = blockIdx.x * blockDim.x + threadIdx.x;
  if (i < n) out[i] = f32_to_bf16_bits(in[i]);
}

__global__ void init_barrier(int* bar) {
  if (threadIdx.x < 2) bar[threadIdx.x] = 0;
}

// ---------------- generic WMMA GEMM: C = A(bf16, MxK, lda) @ Bpacked^T + bias1 (+bias2)
// One wave computes a 32x64 output strip (2 m-tiles x 4 n-tiles); K in steps of 32,
// with one-k-step-ahead operand prefetch (wrap prefetch keeps the loop branch-free).
__global__ __launch_bounds__(THREADS)
void wmma_gemm(const unsigned short* __restrict__ A, long lda,
               const unsigned short* __restrict__ Bp,
               const float* __restrict__ bias1, const float* __restrict__ bias2,
               float* __restrict__ C, long ldc,
               int Mstrips, int Nstrips, int KT) {
  int wave = (blockIdx.x * blockDim.x + threadIdx.x) >> 5;
  int lane = threadIdx.x & 31;
  if (wave >= Mstrips * Nstrips) return;
  int ms = wave / Nstrips;
  int nb = (wave - ms * Nstrips) * 4;          // base n-tile
  int kA = (lane & 16) ? 16 : 0;               // A-layout K sub-offset per lane half
  const unsigned short* Arow0 = A + (size_t)(ms * 32 + (lane & 15)) * lda;
  const unsigned short* Arow1 = Arow0 + (size_t)16 * lda;
  size_t tstride = (size_t)KT * 512;           // elements per packed n-tile
  const unsigned short* bb = Bp + (size_t)nb * tstride + (size_t)lane * 16;

  v8f acc[2][4] = {};
  // prefetch kt = 0
  v16bf a0n = load_pair(Arow0 + kA, Arow0 + kA + 8);
  v16bf a1n = load_pair(Arow1 + kA, Arow1 + kA + 8);
  v16bf bn0 = load_pair(bb + 0 * tstride, bb + 0 * tstride + 8);
  v16bf bn1 = load_pair(bb + 1 * tstride, bb + 1 * tstride + 8);
  v16bf bn2 = load_pair(bb + 2 * tstride, bb + 2 * tstride + 8);
  v16bf bn3 = load_pair(bb + 3 * tstride, bb + 3 * tstride + 8);

  for (int kt = 0; kt < KT; ++kt) {
    v16bf a0 = a0n, a1 = a1n;
    v16bf b0 = bn0, b1 = bn1, b2 = bn2, b3 = bn3;
    int ktn = (kt + 1 < KT) ? (kt + 1) : 0;    // wrap prefetch (harmless re-read)
    {
      const unsigned short* ap0 = Arow0 + ktn * 32 + kA;
      const unsigned short* ap1 = Arow1 + ktn * 32 + kA;
      const unsigned short* bp = bb + (size_t)ktn * 512;
      a0n = load_pair(ap0, ap0 + 8);
      a1n = load_pair(ap1, ap1 + 8);
      bn0 = load_pair(bp + 0 * tstride, bp + 0 * tstride + 8);
      bn1 = load_pair(bp + 1 * tstride, bp + 1 * tstride + 8);
      bn2 = load_pair(bp + 2 * tstride, bp + 2 * tstride + 8);
      bn3 = load_pair(bp + 3 * tstride, bp + 3 * tstride + 8);
    }
    acc[0][0] = __builtin_amdgcn_wmma_f32_16x16x32_bf16(false, a0, false, b0, (short)0, acc[0][0], false, false);
    acc[0][1] = __builtin_amdgcn_wmma_f32_16x16x32_bf16(false, a0, false, b1, (short)0, acc[0][1], false, false);
    acc[0][2] = __builtin_amdgcn_wmma_f32_16x16x32_bf16(false, a0, false, b2, (short)0, acc[0][2], false, false);
    acc[0][3] = __builtin_amdgcn_wmma_f32_16x16x32_bf16(false, a0, false, b3, (short)0, acc[0][3], false, false);
    acc[1][0] = __builtin_amdgcn_wmma_f32_16x16x32_bf16(false, a1, false, b0, (short)0, acc[1][0], false, false);
    acc[1][1] = __builtin_amdgcn_wmma_f32_16x16x32_bf16(false, a1, false, b1, (short)0, acc[1][1], false, false);
    acc[1][2] = __builtin_amdgcn_wmma_f32_16x16x32_bf16(false, a1, false, b2, (short)0, acc[1][2], false, false);
    acc[1][3] = __builtin_amdgcn_wmma_f32_16x16x32_bf16(false, a1, false, b3, (short)0, acc[1][3], false, false);
  }

  int colb = nb * 16 + (lane & 15);
  int rhalf = (lane & 16) ? 8 : 0;             // C/D layout: lanes 16-31 hold M+8
#pragma unroll
  for (int h = 0; h < 2; ++h) {
    int rowb = ms * 32 + h * 16 + rhalf;
#pragma unroll
    for (int i = 0; i < 4; ++i) {
      int col = colb + i * 16;
      float badd = bias1[col] + (bias2 ? bias2[col] : 0.0f);
#pragma unroll
      for (int r = 0; r < 8; ++r)
        C[(size_t)(rowb + r) * ldc + col] = acc[h][i][r] + badd;
    }
  }
}

// ---------------- grid-wide barrier (persistent scan kernel)
__device__ __forceinline__ void grid_barrier(int* cnt, int* gen, int nblk) {
  __syncthreads();
  if (threadIdx.x == 0) {
    __threadfence();
    int g = __hip_atomic_load(gen, __ATOMIC_RELAXED, __HIP_MEMORY_SCOPE_AGENT);
    int prev = __hip_atomic_fetch_add(cnt, 1, __ATOMIC_ACQ_REL, __HIP_MEMORY_SCOPE_AGENT);
    if (prev == nblk - 1) {
      __hip_atomic_store(cnt, 0, __ATOMIC_RELAXED, __HIP_MEMORY_SCOPE_AGENT);
      __hip_atomic_store(gen, g + 1, __ATOMIC_RELEASE, __HIP_MEMORY_SCOPE_AGENT);
    } else {
      while (__hip_atomic_load(gen, __ATOMIC_ACQUIRE, __HIP_MEMORY_SCOPE_AGENT) == g) {
        __builtin_amdgcn_s_sleep(2);
      }
    }
  }
  __syncthreads();
}

// ---------------- persistent recurrent scan: h_t = tanh(pre_t + h_{t-1} @ Whh^T)
// 256 waves (32 blocks x 8 waves): wave -> (mt 0..3, nt 0..63).
// Each wave stages its 32 KB Whh strip into LDS ONCE via async global->LDS DMA
// (8 waves x 32 KB = 256 KB/WGP), then all T steps read B operands with ds_load_b128.
// SEQ=true : read/write the full (B,T,H) bf16 sequence (layer 0).
// SEQ=false: ping-pong (B,H) bf16 buffers; final h lands in buf0 (T even).
template <bool SEQ>
__global__ __launch_bounds__(THREADS)
void rnn_scan(const float* pre, const unsigned short* __restrict__ Wp,
              unsigned short* seq, unsigned short* buf0, unsigned short* buf1,
              int* bar, int T, int nblk) {
  extern __shared__ unsigned short lds[];
  const int KT = HH >> 5;                      // 32 k-steps of 32
  int wl = threadIdx.x >> 5;                   // wave-in-block 0..7
  int wave = (blockIdx.x << 3) | wl;
  int lane = threadIdx.x & 31;
  int mt = wave >> 6;                          // 0..3  (batch tiles, B=64)
  int nt = wave & 63;                          // 0..63 (H tiles)
  int m = mt * 16 + (lane & 15);
  int kA = (lane & 16) ? 16 : 0;
  int colb = nt * 16 + (lane & 15);
  int rb = mt * 16 + ((lane & 16) ? 8 : 0);
  unsigned short* bufs[2] = {buf0, buf1};

  // stage this wave's Whh strip (KT tiles of n-tile nt) into LDS via async DMA
  unsigned short* lw = lds + (size_t)wl * (KT * 512) + (size_t)lane * 16;
  unsigned int lw_off = (unsigned int)(uintptr_t)lw;     // LDS byte address
  const unsigned short* gw = Wp + (size_t)nt * (KT * 512) + (size_t)lane * 16;
  for (int kt = 0; kt < KT; ++kt) {
    async_b128_to_lds(lw_off + kt * 1024,      gw + kt * 512);
    async_b128_to_lds(lw_off + kt * 1024 + 16, gw + kt * 512 + 8);
  }
  wait_async0();
  __syncthreads();

  for (int t = 0; t < T; ++t) {
    // prefetch the pre-activation addends early (overlaps the WMMA chain)
    float padd[8];
#pragma unroll
    for (int r = 0; r < 8; ++r)
      padd[r] = pre[((size_t)(rb + r) * T + (size_t)t) * HH + colb];

    v8f acc = {};
    if (t > 0) {
      const unsigned short* Arow =
          SEQ ? (seq + ((size_t)m * T + (size_t)(t - 1)) * HH)
              : (bufs[t & 1] + (size_t)m * HH);
      // one-ahead A prefetch; B comes from LDS
      v16bf an = load_pair(Arow + kA, Arow + kA + 8);
      const unsigned short* lwk = lw;
#pragma unroll 4
      for (int kt = 0; kt < KT; ++kt) {
        v16bf a = an;
        int ktn = (kt + 1 < KT) ? (kt + 1) : 0;   // wrap prefetch
        const unsigned short* apn = Arow + ktn * 32 + kA;
        an = load_pair(apn, apn + 8);
        v16bf b = load_pair(lwk, lwk + 8);
        acc = __builtin_amdgcn_wmma_f32_16x16x32_bf16(false, a, false, b, (short)0,
                                                      acc, false, false);
        lwk += 512;
      }
    }
#pragma unroll
    for (int r = 0; r < 8; ++r) {
      float v = tanhf(acc[r] + padd[r]);
      unsigned short hv = f32_to_bf16_bits(v);
      if (SEQ) seq[((size_t)(rb + r) * T + (size_t)t) * HH + colb] = hv;
      else     bufs[(t + 1) & 1][(size_t)(rb + r) * HH + colb] = hv;
    }
    grid_barrier(bar, bar + 1, nblk);
  }
}

extern "C" void kernel_launch(void* const* d_in, const int* in_sizes, int n_in,
                              void* d_out, int out_size, void* d_ws, size_t ws_size,
                              hipStream_t stream) {
  (void)in_sizes; (void)n_in; (void)out_size; (void)ws_size;
  const float* x     = (const float*)d_in[0];
  const float* W_ih0 = (const float*)d_in[1];
  const float* W_hh0 = (const float*)d_in[2];
  const float* b_ih0 = (const float*)d_in[3];
  const float* b_hh0 = (const float*)d_in[4];
  const float* W_ih1 = (const float*)d_in[5];
  const float* W_hh1 = (const float*)d_in[6];
  const float* b_ih1 = (const float*)d_in[7];
  const float* b_hh1 = (const float*)d_in[8];
  const float* W_fc  = (const float*)d_in[9];
  const float* b_fc  = (const float*)d_in[10];
  float* out = (float*)d_out;

  // workspace carve-up
  size_t off = 0;
  auto alloc = [&](size_t bytes) {
    void* p = (char*)d_ws + off;
    off += (bytes + 255) & ~(size_t)255;
    return p;
  };
  float*          pre   = (float*)alloc((size_t)BB * TT * HH * 4);   // reused layer0/layer1
  unsigned short* hs    = (unsigned short*)alloc((size_t)BB * TT * HH * 2);
  unsigned short* xb    = (unsigned short*)alloc((size_t)BB * TT * DIN * 2);
  unsigned short* wih0p = (unsigned short*)alloc((size_t)HH * DIN * 2);
  unsigned short* whh0p = (unsigned short*)alloc((size_t)HH * HH * 2);
  unsigned short* wih1p = (unsigned short*)alloc((size_t)HH * HH * 2);
  unsigned short* whh1p = (unsigned short*)alloc((size_t)HH * HH * 2);
  unsigned short* wfcp  = (unsigned short*)alloc((size_t)DOUT * HH * 2);
  unsigned short* hbuf0 = (unsigned short*)alloc((size_t)BB * HH * 2);
  unsigned short* hbuf1 = (unsigned short*)alloc((size_t)BB * HH * 2);
  int*            bar   = (int*)alloc(256);

  auto pack_blocks = [](int N, int K) { return ((N >> 4) * (K >> 5) * 32 + THREADS - 1) / THREADS; };

  // 1) pack weights into bf16 WMMA B-tiles
  pack_weight_bf16<<<pack_blocks(HH, DIN), THREADS, 0, stream>>>(W_ih0, wih0p, HH, DIN);
  pack_weight_bf16<<<pack_blocks(HH, HH),  THREADS, 0, stream>>>(W_hh0, whh0p, HH, HH);
  pack_weight_bf16<<<pack_blocks(HH, HH),  THREADS, 0, stream>>>(W_ih1, wih1p, HH, HH);
  pack_weight_bf16<<<pack_blocks(HH, HH),  THREADS, 0, stream>>>(W_hh1, whh1p, HH, HH);
  pack_weight_bf16<<<pack_blocks(DOUT, HH), THREADS, 0, stream>>>(W_fc, wfcp, DOUT, HH);

  // 2) x -> bf16
  {
    int n = BB * TT * DIN;
    cvt_f32_bf16<<<(n + THREADS - 1) / THREADS, THREADS, 0, stream>>>(x, xb, n);
  }

  // 3) barrier init (deterministic each launch)
  init_barrier<<<1, 32, 0, stream>>>(bar);

  const size_t scan_lds = (size_t)8 * (HH >> 5) * 512 * 2;   // 256 KB

  // 4) layer-0 input projection: pre = x @ W_ih0^T + b_ih0 + b_hh0   (M=32768,N=1024,K=512)
  {
    int Mstrips = (BB * TT) >> 5, Nstrips = HH >> 6, KT = DIN >> 5;
    int blocks = (Mstrips * Nstrips * 32 + THREADS - 1) / THREADS;
    wmma_gemm<<<blocks, THREADS, 0, stream>>>(xb, (long)DIN, wih0p, b_ih0, b_hh0,
                                              pre, (long)HH, Mstrips, Nstrips, KT);
  }

  // 5) layer-0 recurrent scan (persistent, 32 blocks, LDS-resident weights)
  rnn_scan<true><<<32, THREADS, scan_lds, stream>>>(pre, whh0p, hs, nullptr, nullptr, bar, TT, 32);

  // 6) layer-1 input projection: pre = h1 @ W_ih1^T + b_ih1 + b_hh1  (K=1024)
  {
    int Mstrips = (BB * TT) >> 5, Nstrips = HH >> 6, KT = HH >> 5;
    int blocks = (Mstrips * Nstrips * 32 + THREADS - 1) / THREADS;
    wmma_gemm<<<blocks, THREADS, 0, stream>>>(hs, (long)HH, wih1p, b_ih1, b_hh1,
                                              pre, (long)HH, Mstrips, Nstrips, KT);
  }

  // 7) layer-1 recurrent scan (ping-pong h; T even -> final h in hbuf0)
  rnn_scan<false><<<32, THREADS, scan_lds, stream>>>(pre, whh1p, nullptr, hbuf0, hbuf1, bar, TT, 32);

  // 8) FC: out = h_last @ W_fc^T + b_fc   (M=64,N=512,K=1024)
  {
    int Mstrips = BB >> 5, Nstrips = DOUT >> 6, KT = HH >> 5;
    int blocks = (Mstrips * Nstrips * 32 + THREADS - 1) / THREADS;
    wmma_gemm<<<blocks, THREADS, 0, stream>>>(hbuf0, (long)HH, wfcp, b_fc, nullptr,
                                              out, (long)DOUT, Mstrips, Nstrips, KT);
  }
}
